// S4DBlock_18880676233913
// MI455X (gfx1250) — matrix operand order repaired and checked by reading, not compiled, not run on hardware
//
#include <hip/hip_runtime.h>
#include <hip/hip_bf16.h>
#include <stdint.h>

// Problem sizes (fixed by the reference)
constexpr int Bsz  = 16;    // batch
constexpr int Lseq = 4096;  // sequence length
constexpr int Dch  = 256;   // channels
constexpr int Nst  = 64;    // state size
constexpr int Tc   = 64;    // chunk length
constexpr int NCH  = Lseq / Tc;
#define DT_HALF 0.05f

typedef __attribute__((ext_vector_type(16))) __bf16          v16bf;
typedef __attribute__((ext_vector_type(16))) unsigned short  v16us;
typedef __attribute__((ext_vector_type(8)))  unsigned short  v8us;
typedef __attribute__((ext_vector_type(8)))  float           v8f;
typedef __attribute__((ext_vector_type(4)))  unsigned int    v4u;
typedef __attribute__((ext_vector_type(8)))  int             v8i;
typedef __attribute__((ext_vector_type(4)))  int             v4i;

__device__ __forceinline__ unsigned short f2bf(float f) {
    unsigned u = __builtin_bit_cast(unsigned, f);
    u += 0x7FFFu + ((u >> 16) & 1u);   // round-to-nearest-even
    return (unsigned short)(u >> 16);
}

struct SharedBlk {
    float pw[Tc + 1][Nst];              // a^t, t = 0..64
    float bL[Nst], cL[Nst], cbL[Nst], kv[Tc], decay[Nst];
    unsigned short Kbf[Tc * Tc];        // Toeplitz intra-chunk kernel [t][s]
    unsigned short Ebf[Tc * Nst];       // state->output              [t][n]
    unsigned short Gbf[Nst * Tc];       // input->state               [n][s]
    alignas(32) unsigned short XcSw[2 * 32 * 16];  // X chunk, WMMA-B swizzled
    alignas(32) unsigned short H0Sw[2 * 32 * 16];  // H0 bf16, WMMA-B swizzled
    alignas(16) float Xstage[2][Bsz * Tc];         // TDM staging: [b][t] f32
};

// ---- Tensor Data Mover: DMA one x-chunk (tile 1 x 64(t) x 16(b)) into LDS.
// D# per CDNA5 ISA: group0 {count,lds,global,type},
// group1 {data_size=4B, dims, tile dims, dim strides}, groups 2/3 pad to 3D.
// This toolchain's builtin takes 6 args: (g0, g1, g2, g3, extra, cpol).
__device__ __forceinline__ void tdm_issue_chunk(const float* gptr, unsigned lds_off) {
#if __has_builtin(__builtin_amdgcn_tensor_load_to_lds)
    uint64_t ga = (uint64_t)(uintptr_t)gptr;
    v4u g0; v8i g1; v4i g2, g3; v8i g4;
    g0[0] = 1u;                                  // count=1, user descriptor
    g0[1] = lds_off;                             // lds_addr (bytes)
    g0[2] = (unsigned)ga;                        // global_addr[31:0]
    g0[3] = ((unsigned)(ga >> 32) & 0x1FFFFFFu)  // global_addr[56:32]
          | (2u << 30);                          // type=2 ("image")
    g1[0] = (int)0x00020000u;                    // data_size=2 (4 bytes)
    g1[1] = (int)0x01000000u;                    // tensor_dim0 = 256 (low16=0|hi<<16)
    g1[2] = 0;                                   // tensor_dim1 = 1<<20 (low16 here = 0)
    g1[3] = (int)0x00010010u;                    // tensor_dim1 hi | tile_dim0=1 << 16
    g1[4] = (int)0x00100040u;                    // tile_dim1=64 | tile_dim2=16 << 16
    g1[5] = (int)(unsigned)Dch;                  // tensor_dim0_stride = 256 (t step)
    g1[6] = 0;                                   // stride0 hi | stride1 low16 (1M -> 0)
    g1[7] = (int)((unsigned)(Lseq * Dch) >> 16); // tensor_dim1_stride hi (b step = L*D)
    g2[0] = (int)0x00100000u;                    // tensor_dim2 = 1<<20
    g2[1] = 1;                                   // tensor_dim3 = 1
    g2[2] = 0;                                   // tensor_dim2_stride lo
    g2[3] = 0;                                   // stride hi | tile_dim3=0
    g3[0] = 0; g3[1] = (int)0x00010000u;         // tensor_dim4 = 1
    g3[2] = 0; g3[3] = 0;
    g4[0] = 0; g4[1] = 0; g4[2] = 0; g4[3] = 0;  // extra group: zero-filled
    g4[4] = 0; g4[5] = 0; g4[6] = 0; g4[7] = 0;
    __builtin_amdgcn_tensor_load_to_lds(g0, g1, g2, g3, g4, 0 /*cpol*/);
#else
    (void)gptr; (void)lds_off;
#endif
}

__device__ __forceinline__ void tdm_wait(int allow_outstanding) {
#if __has_builtin(__builtin_amdgcn_tensor_load_to_lds)
    if (allow_outstanding) __builtin_amdgcn_s_wait_tensorcnt(1);
    else                   __builtin_amdgcn_s_wait_tensorcnt(0);
#else
    (void)allow_outstanding;
#endif
}

// A-operand (16x32 bf16) gather in contiguous k order, k = kb + 16*(lane/16) + s (prologue only).
__device__ __forceinline__ v16bf loadA16(const unsigned short* M, int mr, int kb, int lane) {
    v16us u;
    const int g = lane >> 4, row = mr + (lane & 15);
#pragma unroll
    for (int j = 0; j < 8; ++j) {
        int k = kb + 2 * j + 16 * g;
        u[2 * j]     = M[row * 64 + k];
        u[2 * j + 1] = M[row * 64 + k + 1];
    }
    return __builtin_bit_cast(v16bf, u);
}

// B-operand from swizzled LDS: one lane = 32 contiguous bytes (2x ds_load_b128).
__device__ __forceinline__ v16bf loadBswz(const unsigned short* Msw, int tile, int lane) {
    const v16us* p = (const v16us*)(Msw + ((tile * 32) + lane) * 16);
    return __builtin_bit_cast(v16bf, *p);
}

__global__ __launch_bounds__(128) void s4d_chunked_wmma(
    const float* __restrict__ x,   // [B, L, D]
    const float* __restrict__ A,   // [D, N]
    const float* __restrict__ Bm,  // [D, N]
    const float* __restrict__ Cm,  // [D, N]
    float* __restrict__ out)       // [B, L, D]
{
    __shared__ SharedBlk S;
    const int d    = blockIdx.x;     // one channel per block
    const int tid  = threadIdx.x;    // 128 threads = 4 waves (wave32)
    const int lane = tid & 31;
    const int w    = tid >> 5;       // wave id -> M-tile row block

    // Kick off the async DMA for chunks 0 and 1 immediately; they overlap the
    // entire prologue computation below. (One TDM issue per block: wave 0.)
    const bool have_tdm =
#if __has_builtin(__builtin_amdgcn_tensor_load_to_lds)
        true;
#else
        false;
#endif
    if (have_tdm && w == 0) {
        tdm_issue_chunk(x + d,            (unsigned)(uintptr_t)&S.Xstage[0][0]);
        tdm_issue_chunk(x + d + Tc * Dch, (unsigned)(uintptr_t)&S.Xstage[1][0]);
    }

    // ---- Prologue: bilinear discretization + powers of a_bar --------------
    if (tid < Nst) {
        float a = A[d * Nst + tid], b = Bm[d * Nst + tid], c = Cm[d * Nst + tid];
        float denom = 1.0f - a * DT_HALF;
        float abar  = (1.0f + a * DT_HALF) / denom;
        float bbar  = (2.0f * DT_HALF / denom) * b;
        S.bL[tid] = bbar; S.cL[tid] = c; S.cbL[tid] = c * bbar;
        float p = 1.0f;
        for (int t = 0; t <= Tc; ++t) { S.pw[t][tid] = p; p *= abar; }
        S.decay[tid] = S.pw[Tc][tid];
    }
    __syncthreads();

    if (tid < Tc) {                       // kv[tau] = sum_n c b a^tau
        float s = 0.0f;
        for (int n = 0; n < Nst; ++n) s += S.cbL[n] * S.pw[tid][n];
        S.kv[tid] = s;
    }
    __syncthreads();

    for (int i = tid; i < Tc * Tc; i += 128) {   // per-channel bf16 matrices
        int r = i >> 6, q = i & 63;
        S.Ebf[i] = f2bf(S.cL[q] * S.pw[r + 1][q]);   // E[r][n=q] = c a^{r+1}
        S.Gbf[i] = f2bf(S.bL[r] * S.pw[63 - q][r]);  // G[n=r][q] = b a^{63-q}
        S.Kbf[i] = (r >= q) ? f2bf(S.kv[r - q]) : (unsigned short)0;
    }
    {   // zero the swizzled H0 buffer
        v8us z = {0,0,0,0,0,0,0,0};
        for (int i = tid; i < 128; i += 128) {
            *(v8us*)(S.H0Sw + i * 8) = z;
            *(v8us*)(S.H0Sw + 1024 + i * 8) = z;
        }
    }
    __syncthreads();

    // ---- Preload constant A-operands for this wave's M-tile ---------------
    const int mr = 16 * w;
    const v16bf Ka0 = loadA16(S.Kbf, mr,  0, lane);
    const v16bf Ka1 = loadA16(S.Kbf, mr, 32, lane);
    const v16bf Ea0 = loadA16(S.Ebf, mr,  0, lane);
    const v16bf Ea1 = loadA16(S.Ebf, mr, 32, lane);
    const v16bf Ga0 = loadA16(S.Gbf, mr,  0, lane);
    const v16bf Ga1 = loadA16(S.Gbf, mr, 32, lane);

    const int g = lane >> 4, bcol = lane & 15;
    const int n0 = mr + 8 * g;            // this lane's first state row

    float dec[8], hstate[8];
#pragma unroll
    for (int v = 0; v < 8; ++v) { dec[v] = S.decay[n0 + v]; hstate[v] = 0.0f; }

    // Convert-phase indices: each thread owns 8 consecutive t for one b.
    const int cb = tid >> 3;              // batch column 0..15
    const int tb = (tid & 7) * 8;         // t base 0..56
    const int ctile = tb >> 5, clh = (tb >> 4) & 1;
    unsigned short* const xsw_dst =
        S.XcSw + ((ctile * 32) + (cb + 16 * clh)) * 16 + (tb & 15);

    // ---- Sequential chunk loop --------------------------------------------
    for (int c = 0; c < NCH; ++c) {
        const int t0 = c * Tc;

        // (1) current chunk's DMA done? (one more may stay in flight)
        if (have_tdm) {
            if (w == 0) tdm_wait(c + 1 < NCH ? 1 : 0);
        } else if (w == 0) {
            // fallback staging path: plain strided gather into Xstage
            for (int i = lane; i < Bsz * Tc; i += 32) {
                int b = i >> 6, r = i & 63;
                S.Xstage[c & 1][b * 64 + r] =
                    x[((size_t)b * Lseq + (size_t)(t0 + r)) * Dch + d];
            }
        }
        __syncthreads();

        // (2) f32 -> bf16, scatter into WMMA-B swizzled layout (1x b128 store)
        {
            const float* src = S.Xstage[c & 1] + cb * 64 + tb;
            float4 xa = *(const float4*)(src);
            float4 xb = *(const float4*)(src + 4);
            v8us pk;
            pk[0] = f2bf(xa.x); pk[1] = f2bf(xa.y); pk[2] = f2bf(xa.z); pk[3] = f2bf(xa.w);
            pk[4] = f2bf(xb.x); pk[5] = f2bf(xb.y); pk[6] = f2bf(xb.z); pk[7] = f2bf(xb.w);
            *(v8us*)xsw_dst = pk;
        }
        __syncthreads();

        // (3) staging buffer (c&1) is free again -> prefetch chunk c+2 via TDM
        if (have_tdm && w == 0 && c + 2 < NCH)
            tdm_issue_chunk(x + d + (size_t)(t0 + 2 * Tc) * Dch,
                            (unsigned)(uintptr_t)&S.Xstage[c & 1][0]);

        // (4) WMMA: Y = K*X + E*H0 ; Hnew = G*X
        const v16bf BX0 = loadBswz(S.XcSw, 0, lane);
        const v16bf BX1 = loadBswz(S.XcSw, 1, lane);
        const v16bf BH0 = loadBswz(S.H0Sw, 0, lane);
        const v16bf BH1 = loadBswz(S.H0Sw, 1, lane);

        v8f yacc = {0.f,0.f,0.f,0.f,0.f,0.f,0.f,0.f};
        yacc = __builtin_amdgcn_wmma_f32_16x16x32_bf16(false, Ka0, false, BX0, (short)0, yacc, false, false);
        yacc = __builtin_amdgcn_wmma_f32_16x16x32_bf16(false, Ka1, false, BX1, (short)0, yacc, false, false);
        yacc = __builtin_amdgcn_wmma_f32_16x16x32_bf16(false, Ea0, false, BH0, (short)0, yacc, false, false);
        yacc = __builtin_amdgcn_wmma_f32_16x16x32_bf16(false, Ea1, false, BH1, (short)0, yacc, false, false);

        v8f hacc = {0.f,0.f,0.f,0.f,0.f,0.f,0.f,0.f};
        hacc = __builtin_amdgcn_wmma_f32_16x16x32_bf16(false, Ga0, false, BX0, (short)0, hacc, false, false);
        hacc = __builtin_amdgcn_wmma_f32_16x16x32_bf16(false, Ga1, false, BX1, (short)0, hacc, false, false);

        // Y out + state update in registers (state lives per-lane in VGPRs)
        v8us hp;
#pragma unroll
        for (int v = 0; v < 8; ++v) {
            int r = n0 + v;   // D-matrix row M = v + 8*(lane>=16), + wave tile
            out[((size_t)bcol * Lseq + (size_t)(t0 + r)) * Dch + d] = yacc[v];
            hstate[v] = dec[v] * hstate[v] + hacc[v];
            hp[v] = f2bf(hstate[v]);
        }
        __syncthreads();   // all waves done reading H0Sw/XcSw

        // (5) publish new H0 (bf16, swizzled): one 16B LDS store per lane
        *(v8us*)(S.H0Sw + (((n0 >> 5) * 32) + (bcol + 16 * ((n0 >> 4) & 1))) * 16
                 + (n0 & 15)) = hp;
        // next iteration's barriers order this write before the next reads
    }
}

extern "C" void kernel_launch(void* const* d_in, const int* in_sizes, int n_in,
                              void* d_out, int out_size, void* d_ws, size_t ws_size,
                              hipStream_t stream) {
    const float* x  = (const float*)d_in[0];
    const float* A  = (const float*)d_in[1];
    const float* Bm = (const float*)d_in[2];
    const float* Cm = (const float*)d_in[3];
    float* out = (float*)d_out;
    s4d_chunked_wmma<<<dim3(Dch), dim3(128), 0, stream>>>(x, A, Bm, Cm, out);
}
